// LoopHead_89773406421561
// MI455X (gfx1250) — matrix operand order repaired and checked by reading, not compiled, not run on hardware
//
#include <hip/hip_runtime.h>
#include <hip/hip_bf16.h>

typedef __attribute__((ext_vector_type(16))) _Float16 v16h;
typedef __attribute__((ext_vector_type(8)))  _Float16 v8h;
typedef __attribute__((ext_vector_type(8)))  float    v8f;
typedef __attribute__((ext_vector_type(4)))  float    v4f;

#define IN_Z   32
#define PROJ_H 64
#define LOOP_D 32
#define DEC_H  64

// D = A*B + C, f16 inputs, f32 accum, 16x16x32
__device__ __forceinline__ v8f wmma_f16(v16h a, v16h b, v8f c) {
    return __builtin_amdgcn_wmma_f32_16x16x32_f16(
        /*neg_a=*/false, a, /*neg_b=*/false, b,
        /*c_mod=*/(short)0, c, /*reuse_a=*/false, /*reuse_b=*/false);
}

// Build one 16x32 A-tile of W^T (A[m][k] = W[kbase+k][row]) in contiguous k order (B uses the same):
// lane<16: elems 0..15 -> K 0..15
// lane>=16: elems 0..15 -> K 16..31
__device__ __forceinline__ v16h load_wT_tile(const float* __restrict__ W,
                                             int ncols, int row, int kbase, int hk) {
    v16h a;
#pragma unroll
    for (int e = 0; e < 16; ++e) {
        int k = kbase + (hk << 4) + e;
        a[e] = (_Float16)W[k * ncols + row];
    }
    return a;
}

// ---------------- Kernel 1: per-node MLP 32 -> 64(relu) -> 32, output f16 ----
// Computes transposed GEMMs: H1^T = w1^T @ zmod^T ; zloop^T = w2^T @ relu(H1)^T
__global__ void __launch_bounds__(256)
node_proj_kernel(const float* __restrict__ z_mod,
                 const float* __restrict__ w1, const float* __restrict__ b1,
                 const float* __restrict__ w2, const float* __restrict__ b2,
                 _Float16* __restrict__ zloop, int nNodes) {
    __shared__ __align__(32) _Float16 hbuf[8][16 * PROJ_H];  // 2KB per wave
    const int lane = threadIdx.x & 31;
    const int wave = threadIdx.x >> 5;
    const int hk   = lane >> 4;     // lane half
    const int mrow = lane & 15;
    const int nwaves = (gridDim.x * blockDim.x) >> 5;
    const int gw     = (blockIdx.x * blockDim.x + threadIdx.x) >> 5;

    // Wave-invariant weight tiles in registers
    v16h A1[4];          // w1^T: 4 feature tiles, K=32
    v16h A2[2][2];       // w2^T: 2 dim tiles x 2 K-tiles (K=64)
#pragma unroll
    for (int ft = 0; ft < 4; ++ft) A1[ft] = load_wT_tile(w1, PROJ_H, ft * 16 + mrow, 0, hk);
#pragma unroll
    for (int dt = 0; dt < 2; ++dt)
#pragma unroll
        for (int kt = 0; kt < 2; ++kt)
            A2[dt][kt] = load_wT_tile(w2, LOOP_D, dt * 16 + mrow, kt * 32, hk);
    v8f bias1[4], bias2[2];
#pragma unroll
    for (int ft = 0; ft < 4; ++ft)
#pragma unroll
        for (int r = 0; r < 8; ++r) bias1[ft][r] = b1[ft * 16 + hk * 8 + r];
#pragma unroll
    for (int dt = 0; dt < 2; ++dt)
#pragma unroll
        for (int r = 0; r < 8; ++r) bias2[dt][r] = b2[dt * 16 + hk * 8 + r];

    const int ngroups = (nNodes + 15) >> 4;
    for (int g = gw; g < ngroups; g += nwaves) {
        const int node   = g * 16 + mrow;
        const int nclamp = node < nNodes ? node : nNodes - 1;

        // B operand = zmod^T: 16 consecutive f32 dims (hk*16 ..) of this node, cvt f16
        const v4f* zv = (const v4f*)(z_mod + (long)nclamp * IN_Z + hk * 16);
        v4f f0 = zv[0], f1 = zv[1], f2 = zv[2], f3 = zv[3];
        v16h bin;
#pragma unroll
        for (int j = 0; j < 4; ++j) {
            bin[j]      = (_Float16)f0[j];
            bin[4 + j]  = (_Float16)f1[j];
            bin[8 + j]  = (_Float16)f2[j];
            bin[12 + j] = (_Float16)f3[j];
        }

        // Layer 1: 4 WMMAs -> H1^T tiles; relu; pack row-major [node][f] into LDS
#pragma unroll
        for (int ft = 0; ft < 4; ++ft) {
            v8f c = bias1[ft];
            c = wmma_f16(A1[ft], bin, c);
            v8h p;
#pragma unroll
            for (int r = 0; r < 8; ++r) p[r] = (_Float16)fmaxf(c[r], 0.f);
            // lane holds node=mrow, features ft*16+hk*8 .. +7 -> contiguous b128 store
            *(v8h*)&hbuf[wave][mrow * PROJ_H + ft * 16 + hk * 8] = p;
        }
        // Same-wave DS ops are in-order: no barrier needed.
        v16h b2t0 = *(const v16h*)&hbuf[wave][mrow * PROJ_H + 0  + hk * 16];
        v16h b2t1 = *(const v16h*)&hbuf[wave][mrow * PROJ_H + 32 + hk * 16];

        // Layer 2: 4 WMMAs -> zloop^T tiles; store f16 row-major [node][dim]
#pragma unroll
        for (int dt = 0; dt < 2; ++dt) {
            v8f c = bias2[dt];
            c = wmma_f16(A2[dt][0], b2t0, c);
            c = wmma_f16(A2[dt][1], b2t1, c);
            v8h p;
#pragma unroll
            for (int r = 0; r < 8; ++r) p[r] = (_Float16)c[r];
            if (node < nNodes)
                *(v8h*)(zloop + (long)node * LOOP_D + dt * 16 + hk * 8) = p;
        }
    }
}

// ---------------- Kernel 2: edge decoder, 16 edges per wave-group ------------
// H^T = dw1^T @ [zi|zj]^T  (8 WMMAs), then per-lane relu-dot with dw2 + shfl.
__global__ void __launch_bounds__(256)
edge_dec_kernel(const _Float16* __restrict__ zloop,
                const int* __restrict__ ei, long long nE,
                const float* __restrict__ dw1, const float* __restrict__ db1,
                const float* __restrict__ dw2, const float* __restrict__ db2,
                float* __restrict__ out) {
    const int lane = threadIdx.x & 31;
    const int hk   = lane >> 4;
    const int mrow = lane & 15;
    const int nwaves = (gridDim.x * blockDim.x) >> 5;
    const int gw     = (blockIdx.x * blockDim.x + threadIdx.x) >> 5;

    // dw1^T tiles: 4 hidden-feature tiles x 2 K-tiles (K=64: zi dims then zj dims)
    v16h A1[4][2];
#pragma unroll
    for (int ft = 0; ft < 4; ++ft)
#pragma unroll
        for (int kt = 0; kt < 2; ++kt)
            A1[ft][kt] = load_wT_tile(dw1, DEC_H, ft * 16 + mrow, kt * 32, hk);
    v8f bias[4], w2r[4];
#pragma unroll
    for (int ft = 0; ft < 4; ++ft)
#pragma unroll
        for (int r = 0; r < 8; ++r) {
            bias[ft][r] = db1[ft * 16 + hk * 8 + r];
            w2r[ft][r]  = dw2[ft * 16 + hk * 8 + r];
        }
    const float db2v = db2[0];

    const long long ngroups = (nE + 15) >> 4;
    for (long long g = gw; g < ngroups; g += nwaves) {
        long long e  = g * 16 + mrow;
        long long ec = e < nE ? e : nE - 1;
        const long long src = ei[ec];
        const long long dst = ei[nE + ec];
        // B operands: 16 consecutive f16 dims of gathered node rows (32B, L2-resident)
        const v16h bsrc = *(const v16h*)(zloop + src * LOOP_D + hk * 16);
        const v16h bdst = *(const v16h*)(zloop + dst * LOOP_D + hk * 16);

        float acc = 0.f;
#pragma unroll
        for (int ft = 0; ft < 4; ++ft) {
            v8f c = bias[ft];
            c = wmma_f16(A1[ft][0], bsrc, c);   // zi part (dw1 rows 0..31)
            c = wmma_f16(A1[ft][1], bdst, c);   // zj part (dw1 rows 32..63)
            // D-tile: this lane holds features ft*16+hk*8+r of edge mrow
#pragma unroll
            for (int r = 0; r < 8; ++r) {
                acc = fmaf(fmaxf(c[r], 0.f), w2r[ft][r], acc);  // co-executes with WMMA
            }
        }
        acc += __shfl_xor(acc, 16, 32);          // combine the two feature halves
        if (hk == 0 && e < nE) out[e] = acc + db2v;
    }
}

extern "C" void kernel_launch(void* const* d_in, const int* in_sizes, int n_in,
                              void* d_out, int out_size, void* d_ws, size_t ws_size,
                              hipStream_t stream) {
    const float* z_mod = (const float*)d_in[0];
    const int*   ei    = (const int*)d_in[1];   // jax default x64-off -> int32
    const float* w1    = (const float*)d_in[2];
    const float* b1    = (const float*)d_in[3];
    const float* w2    = (const float*)d_in[4];
    const float* b2    = (const float*)d_in[5];
    const float* dw1   = (const float*)d_in[6];
    const float* db1   = (const float*)d_in[7];
    const float* dw2   = (const float*)d_in[8];
    const float* db2   = (const float*)d_in[9];

    const int       nNodes = in_sizes[0] / IN_Z;
    const long long nE     = in_sizes[1] / 2;

    _Float16* zloop = (_Float16*)d_ws;          // f16 [nNodes][32], ~6.4MB, L2-resident

    const int ng1     = (nNodes + 15) / 16;
    const int blocks1 = (ng1 + 7) / 8;          // 8 waves per 256-thread block
    node_proj_kernel<<<blocks1, 256, 0, stream>>>(z_mod, w1, b1, w2, b2, zloop, nNodes);

    edge_dec_kernel<<<2048, 256, 0, stream>>>(zloop, ei, nE, dw1, db1, dw2, db2,
                                              (float*)d_out);
}
